// AttentionMM_48155173323181
// MI455X (gfx1250) — matrix operand order repaired and checked by reading, not compiled, not run on hardware
//
#include <hip/hip_runtime.h>
#include <hip/hip_bf16.h>

#define Bc 8
#define Tc 2048
#define Ec 256

typedef __attribute__((ext_vector_type(16))) __bf16 v16bf;
typedef __attribute__((ext_vector_type(8)))  float  v8f;

union AFrag { v16bf v; uint4 q[2]; };

__device__ __forceinline__ float fast_tanh(float x) {
#if __has_builtin(__builtin_amdgcn_tanhf)
    return __builtin_amdgcn_tanhf(x);
#elif __has_builtin(__builtin_amdgcn_tanh_f32)
    return __builtin_amdgcn_tanh_f32(x);
#else
    return tanhf(x);
#endif
}

// ---------------------------------------------------------------------------
// Kernel 1: s1[b,t] = x1[b,t,:].W1 ; s2[b,t] = x2[b,t,:].W2 ; zero at1/at2
// ---------------------------------------------------------------------------
__global__ __launch_bounds__(256) void prep_kernel(
    const float* __restrict__ x1, const float* __restrict__ x2,
    const float* __restrict__ W1, const float* __restrict__ W2,
    float* __restrict__ s1, float* __restrict__ s2,
    float* __restrict__ at1, float* __restrict__ at2)
{
    int w = threadIdx.x >> 5, lane = threadIdx.x & 31;
    int row = blockIdx.x * 8 + w;                 // 0 .. B*T-1
    const float* p1 = x1 + (size_t)row * Ec;
    const float* p2 = x2 + (size_t)row * Ec;
    float a1 = 0.f, a2 = 0.f;
    for (int e = lane; e < Ec; e += 32) {
        a1 += p1[e] * W1[e];
        a2 += p2[e] * W2[e];
    }
    for (int off = 16; off > 0; off >>= 1) {
        a1 += __shfl_xor(a1, off);
        a2 += __shfl_xor(a2, off);
    }
    if (lane == 0) {
        s1[row] = a1; s2[row] = a2;
        at1[row] = 0.f; at2[row] = 0.f;
    }
}

// ---------------------------------------------------------------------------
// Kernel 2: one-shot bf16 conversion of x into two layouts:
//   Xb[b][t][e] (row major)  and  XT[b][e][t] (transposed)
// ---------------------------------------------------------------------------
__global__ __launch_bounds__(256) void cvt_kernel(
    const float* __restrict__ X,
    __bf16* __restrict__ Xb,   // (B,T,E)
    __bf16* __restrict__ XT)   // (B,E,T)
{
    __shared__ __bf16 tile[32][33];
    int b  = blockIdx.z;
    int t0 = blockIdx.x * 32;
    int e0 = blockIdx.y * 32;
    int tid = threadIdx.x;
    int c  = tid & 31;     // column within tile
    int r8 = tid >> 5;     // 0..7
    const float* Xp  = X  + (size_t)b * Tc * Ec;
    __bf16* Xbp = Xb + (size_t)b * Tc * Ec;
    __bf16* XTp = XT + (size_t)b * Tc * Ec;
    #pragma unroll
    for (int rr = 0; rr < 4; ++rr) {
        int r = r8 + rr * 8;                       // t within tile
        __bf16 h = (__bf16)Xp[(size_t)(t0 + r) * Ec + e0 + c];
        Xbp[(size_t)(t0 + r) * Ec + e0 + c] = h;
        tile[r][c] = h;
    }
    __syncthreads();
    #pragma unroll
    for (int rr = 0; rr < 4; ++rr) {
        int e = r8 + rr * 8;                       // e within tile
        XTp[(size_t)(e0 + e) * Tc + t0 + c] = tile[c][e];
    }
}

// ---------------------------------------------------------------------------
// Kernel 3: M[b,i,e] = sum_k tanh(s[b,i]+bias[k]) * X[b,k,e]   (bf16 out)
// block: 64 rows (i) x 256 cols (e). A (tanh tile) staged in LDS once per
// k-step; all 8 B fragments loaded up-front per k-step for deep MLP.
// ---------------------------------------------------------------------------
__global__ __launch_bounds__(256) void tanh_gemm_kernel(
    const __bf16* __restrict__ XT,     // (B,E,T) bf16 transposed
    const float* __restrict__ s,       // (B,T)
    const float* __restrict__ bias,    // (T,)
    __bf16* __restrict__ Mout)         // (B,T,E) bf16
{
    __shared__ __bf16 ldsA[64 * 32];   // [i][k] tanh tile, 4KB

    int b  = blockIdx.y;
    int i0 = blockIdx.x * 64;
    const __bf16* XTb = XT + (size_t)b * Tc * Ec;
    const float*  sb  = s + (size_t)b * Tc;

    int t = threadIdx.x;
    int lane = t & 31, w = t >> 5;
    int itile = w & 3;     // 4 row tiles of 16
    int nh    = w >> 2;    // column half: n-tiles nh*8 .. nh*8+7
    int m = lane & 15, h = lane >> 4;

    // A-generation: thread -> (row ai, 8 consecutive k at ak)
    int ai = t >> 2;
    int ak = (t & 3) * 8;
    float srow = sb[i0 + ai];
    __bf16* ap = &ldsA[ai * 32 + ak];

    // per-lane base for its B column (fixed e per nt), k advances per step
    const __bf16* bcol = XTb + (size_t)((nh * 8) * 16 + m) * Tc + h * 8;

    v8f zero = {0.f,0.f,0.f,0.f,0.f,0.f,0.f,0.f};
    v8f acc[8];
    #pragma unroll
    for (int nt = 0; nt < 8; ++nt) acc[nt] = zero;

    for (int k0 = 0; k0 < Tc; k0 += 32) {
        // stage A: tanh(s[i]+bias[k]) -> ldsA, one b128 store per thread
        {
            const float* bp = bias + k0 + ak;
            union { uint4 q; __bf16 hh[8]; } pk;
            #pragma unroll
            for (int qq = 0; qq < 8; ++qq)
                pk.hh[qq] = (__bf16)fast_tanh(srow + bp[qq]);
            *(uint4*)ap = pk.q;
        }

        // batch-load all 8 B fragments (16 x global_load_b128, clause-able)
        AFrag bf[8];
        #pragma unroll
        for (int nt = 0; nt < 8; ++nt) {
            const __bf16* bb = bcol + (size_t)nt * 16 * Tc + k0;
            bf[nt].q[0] = *(const uint4*)(bb);
            bf[nt].q[1] = *(const uint4*)(bb + 16);
        }

        __syncthreads();
        // A fragment for this wave's i-tile (ISA 16-bit A layout)
        AFrag a;
        {
            const __bf16* base = &ldsA[(itile * 16 + m) * 32];
            a.q[0] = *(const uint4*)(base + h * 8);
            a.q[1] = *(const uint4*)(base + 16 + h * 8);
        }

        // 8 back-to-back WMMAs, no inter-op data hazards
        #pragma unroll
        for (int nt = 0; nt < 8; ++nt)
            acc[nt] = __builtin_amdgcn_wmma_f32_16x16x32_bf16(
                false, a.v, false, bf[nt].v, (short)0, acc[nt], false, false);
        __syncthreads();
    }

    __bf16* Mb = Mout + (size_t)b * Tc * Ec;
    #pragma unroll
    for (int nt = 0; nt < 8; ++nt) {
        int e = (nh * 8 + nt) * 16 + m;
        #pragma unroll
        for (int r = 0; r < 8; ++r) {
            int row = i0 + itile * 16 + h * 8 + r;
            Mb[(size_t)row * Ec + e] = (__bf16)acc[nt][r];
        }
    }
}

// ---------------------------------------------------------------------------
// Kernel 4: score[i,j] = sum_e M[b,i,e]*Xb[b,j,e]; softmax over j; column
// sums over i accumulated into at[b,j] with float atomics. 16 rows/block,
// full 2048-col score strip in LDS (128KB).
// ---------------------------------------------------------------------------
#define ROWS 16

__global__ __launch_bounds__(256) void score_kernel(
    const __bf16* __restrict__ Mmat,  // (B,T,E) bf16
    const __bf16* __restrict__ Xb,    // (B,T,E) bf16 row-major
    float* __restrict__ at)           // (B,T) accumulated
{
    extern __shared__ float sc[];     // ROWS*T scores + m/rd arrays
    float* rowm  = sc + ROWS * Tc;
    float* rowrd = rowm + ROWS;

    int b  = blockIdx.y;
    int i0 = blockIdx.x * ROWS;
    int t = threadIdx.x, lane = t & 31, w = t >> 5;
    int m = lane & 15, h = lane >> 4;

    // all 8 A fragments (K = E = 256) resident in registers
    AFrag a[8];
    const __bf16* Ab = Mmat + ((size_t)b * Tc + i0) * Ec;
    #pragma unroll
    for (int ks = 0; ks < 8; ++ks) {
        const __bf16* p = Ab + (size_t)m * Ec + ks * 32;
        a[ks].q[0] = *(const uint4*)(p + h * 8);
        a[ks].q[1] = *(const uint4*)(p + 16 + h * 8);
    }

    const __bf16* Xbb = Xb + (size_t)b * Tc * Ec;

    // wave w owns j-tiles w, w+8, ...
    for (int jt = w; jt < Tc / 16; jt += 8) {
        const __bf16* xr = Xbb + (size_t)(jt * 16 + m) * Ec + h * 8;

        // batch-load all 8 B fragments for this j-tile
        AFrag bf[8];
        #pragma unroll
        for (int ks = 0; ks < 8; ++ks) {
            const __bf16* p = xr + ks * 32;       // k = 8h..8h+7 and 16+8h..16+8h+7
            bf[ks].q[0] = *(const uint4*)(p);
            bf[ks].q[1] = *(const uint4*)(p + 16);
        }

        v8f acc = {0.f,0.f,0.f,0.f,0.f,0.f,0.f,0.f};
        #pragma unroll
        for (int ks = 0; ks < 8; ++ks)
            acc = __builtin_amdgcn_wmma_f32_16x16x32_bf16(
                false, a[ks].v, false, bf[ks].v, (short)0, acc, false, false);

        #pragma unroll
        for (int r = 0; r < 8; ++r)
            sc[(h * 8 + r) * Tc + jt * 16 + m] = acc[r];
    }
    __syncthreads();

    // per-row max and 1/denominator (wave w -> rows 2w, 2w+1)
    for (int rr = 0; rr < 2; ++rr) {
        int row = w * 2 + rr;
        const float* rp = sc + (size_t)row * Tc;
        float mx = -3.0e38f;
        for (int j = lane; j < Tc; j += 32) mx = fmaxf(mx, rp[j]);
        for (int off = 16; off > 0; off >>= 1) mx = fmaxf(mx, __shfl_xor(mx, off));
        float ds = 0.f;
        for (int j = lane; j < Tc; j += 32) ds += __expf(rp[j] - mx);
        for (int off = 16; off > 0; off >>= 1) ds += __shfl_xor(ds, off);
        if (lane == 0) { rowm[row] = mx; rowrd[row] = 1.0f / ds; }
    }
    __syncthreads();

    float lm[ROWS], lrd[ROWS];
    #pragma unroll
    for (int r = 0; r < ROWS; ++r) { lm[r] = rowm[r]; lrd[r] = rowrd[r]; }

    for (int j = t; j < Tc; j += 256) {
        float colsum = 0.f;
        #pragma unroll
        for (int r = 0; r < ROWS; ++r)
            colsum += __expf(sc[(size_t)r * Tc + j] - lm[r]) * lrd[r];
        atomicAdd(&at[(size_t)b * Tc + j], colsum);
    }
}

// ---------------------------------------------------------------------------
// Kernel 5: o1[b,e] = sum_t x1[b,t,e]*at1[b,t]; out = concat(o1,o2)
// ---------------------------------------------------------------------------
__global__ __launch_bounds__(256) void out_kernel(
    const float* __restrict__ x1, const float* __restrict__ x2,
    const float* __restrict__ at1, const float* __restrict__ at2,
    float* __restrict__ out)
{
    int idx = blockIdx.x * 256 + threadIdx.x;  // 0 .. B*2E-1
    int b = idx / (2 * Ec);
    int c = idx % (2 * Ec);
    const float* x  = (c < Ec) ? x1  : x2;
    const float* at = (c < Ec) ? at1 : at2;
    int e = (c < Ec) ? c : c - Ec;
    const float* xb = x + (size_t)b * Tc * Ec + e;
    const float* ab = at + (size_t)b * Tc;
    float acc = 0.f;
    for (int tt = 0; tt < Tc; ++tt)
        acc += xb[(size_t)tt * Ec] * ab[tt];
    out[idx] = acc;
}

// ---------------------------------------------------------------------------
extern "C" void kernel_launch(void* const* d_in, const int* in_sizes, int n_in,
                              void* d_out, int out_size, void* d_ws, size_t ws_size,
                              hipStream_t stream) {
    (void)in_sizes; (void)n_in; (void)out_size; (void)ws_size;
    const float* x1 = (const float*)d_in[0];
    const float* x2 = (const float*)d_in[1];
    const float* W1 = (const float*)d_in[2];
    const float* b1 = (const float*)d_in[3];
    const float* W2 = (const float*)d_in[4];
    const float* b2 = (const float*)d_in[5];
    float* out = (float*)d_out;

    size_t BT  = (size_t)Bc * Tc;
    size_t BTE = BT * Ec;
    float* s1  = (float*)d_ws;
    float* s2  = s1  + BT;
    float* at1 = s2  + BT;
    float* at2 = at1 + BT;
    __bf16* M1  = (__bf16*)(at2 + BT);
    __bf16* M2  = M1  + BTE;
    __bf16* Xb1 = M2  + BTE;
    __bf16* Xb2 = Xb1 + BTE;
    __bf16* XT1 = Xb2 + BTE;
    __bf16* XT2 = XT1 + BTE;
    // ws usage: 16*BT + 12*BTE bytes ~= 50.6 MB

    prep_kernel<<<(int)(BT / 8), 256, 0, stream>>>(x1, x2, W1, W2, s1, s2, at1, at2);

    dim3 gc(Tc / 32, Ec / 32, Bc);
    cvt_kernel<<<gc, 256, 0, stream>>>(x1, Xb1, XT1);
    cvt_kernel<<<gc, 256, 0, stream>>>(x2, Xb2, XT2);

    dim3 g1(Tc / 64, Bc);
    // M1 = tanh(s2[i]+b2[k]) @ x1 ; M2 = tanh(s1[i]+b1[k]) @ x2
    tanh_gemm_kernel<<<g1, 256, 0, stream>>>(XT1, s2, b2, M1);
    tanh_gemm_kernel<<<g1, 256, 0, stream>>>(XT2, s1, b1, M2);

    dim3 g2(Tc / ROWS, Bc);
    size_t smem = (size_t)ROWS * Tc * 4 + 2 * ROWS * 4;
    // score1 = M1 @ x2^T -> at1 ; score2 = M2 @ x1^T -> at2
    score_kernel<<<g2, 256, smem, stream>>>(M1, Xb2, at1);
    score_kernel<<<g2, 256, smem, stream>>>(M2, Xb1, at2);

    out_kernel<<<(Bc * 2 * Ec) / 256, 256, 0, stream>>>(x1, x2, at1, at2, out);
}
